// SemanticRoutingAttention_64785286693263
// MI455X (gfx1250) — matrix operand order repaired and checked by reading, not compiled, not run on hardware
//
#include <hip/hip_runtime.h>
#include <hip/hip_bf16.h>

#define DEV __device__ __forceinline__

typedef __bf16 v16bf __attribute__((ext_vector_type(16)));
typedef float  v8f   __attribute__((ext_vector_type(8)));
typedef unsigned short u16;

static constexpr int Bb = 32, Nn = 1600, Dd = 768, Gg = 64, Hh = 8, HDd = 96;

union BFrag { v16bf v; uint4 u[2]; };
union Pack8 { uint4 q; u16 s[8]; };

DEV u16 f2bf(float f) {
    unsigned u = __float_as_uint(f);
    u += 0x7FFFu + ((u >> 16) & 1u);          // round-to-nearest-even
    return (u16)(u >> 16);
}
DEV float bf2f(u16 h) { return __uint_as_float(((unsigned)h) << 16); }

// A-matrix 16x32 bf16 fragment (row-major source, ld in elements).
DEV v16bf load_a16x32(const u16* base, int ld, int row0, int k0, int lane) {
    int half = lane >> 4, r = lane & 15;
    const u16* p = base + (size_t)(row0 + r) * ld + k0 + half * 8;
    BFrag f;
    f.u[0] = *(const uint4*)(p);
    f.u[1] = *(const uint4*)(p + 16);
    return f.v;
}
// B-matrix 32x16 bf16 fragment where memory holds B^T row-major.
DEV v16bf load_b32x16(const u16* base, int ld, int col0, int k0, int lane) {
    int half = lane >> 4, r = lane & 15;
    const u16* p = base + (size_t)(col0 + r) * ld + k0 + half * 8;
    BFrag f;
    f.u[0] = *(const uint4*)(p);
    f.u[1] = *(const uint4*)(p + 16);
    return f.v;
}
DEV v8f wmma_bf16(v16bf a, v16bf b, v8f c) {
    return __builtin_amdgcn_wmma_f32_16x16x32_bf16(false, a, false, b, (short)0, c, false, false);
}

// ---------------------------------------------------------------- casts ----
__global__ void cast_f32_bf16(const float* __restrict__ s, u16* __restrict__ d, int n) {
    int i = blockIdx.x * 256 + threadIdx.x;
    if (i < n) d[i] = f2bf(s[i]);
}

// x (B,N,D) f32 -> xbf (B*N,D) bf16 row-major AND xT (B,D,N) bf16
__global__ void prep_x_kernel(const float* __restrict__ x, u16* __restrict__ xbf,
                              u16* __restrict__ xT) {
    __shared__ u16 tile[32][33];
    int bid = blockIdx.x;
    int b = bid / 1200; int rem = bid % 1200;
    int ntile = rem / 24, dt = rem % 24;
    int tx = threadIdx.x & 31, ty = threadIdx.x >> 5;
#pragma unroll
    for (int i = 0; i < 4; ++i) {
        int nl = ty + i * 8;
        size_t idx = ((size_t)b * Nn + ntile * 32 + nl) * Dd + dt * 32 + tx;
        u16 h = f2bf(x[idx]);
        xbf[idx] = h;
        tile[nl][tx] = h;
    }
    __syncthreads();
#pragma unroll
    for (int i = 0; i < 4; ++i) {
        int dl = ty + i * 8;
        xT[((size_t)b * Dd + dt * 32 + dl) * Nn + ntile * 32 + tx] = tile[tx][dl];
    }
}

// ------------------------------------------------------------- routing -----
// R = softmax((x @ W_r^T)/8)  -> written transposed: Rt (B,G,N) bf16
// Ping-pong double-buffered k-loop (24 steps, even).
__global__ __launch_bounds__(128) void routing_kernel(const u16* __restrict__ xbf,
                                                      const u16* __restrict__ wr,
                                                      u16* __restrict__ Rt) {
    int lane = threadIdx.x & 31, wave = threadIdx.x >> 5;
    int idx = blockIdx.x * 4 + wave;
    int row0 = idx * 16;
    int b = row0 / Nn, n0 = row0 % Nn;
    const v8f vz = {0.f,0.f,0.f,0.f,0.f,0.f,0.f,0.f};
    v8f acc[4] = {vz, vz, vz, vz};

    v16bf aA = load_a16x32(xbf, Dd, row0, 0, lane);
    v16bf bA[4];
#pragma unroll
    for (int t = 0; t < 4; ++t) bA[t] = load_b32x16(wr, Dd, t * 16, 0, lane);
    int k = 32;
    for (; k < Dd - 32; k += 64) {
        v16bf aB = load_a16x32(xbf, Dd, row0, k, lane);
        v16bf bB[4];
#pragma unroll
        for (int t = 0; t < 4; ++t) bB[t] = load_b32x16(wr, Dd, t * 16, k, lane);
#pragma unroll
        for (int t = 0; t < 4; ++t) acc[t] = wmma_bf16(aA, bA[t], acc[t]);
        aA = load_a16x32(xbf, Dd, row0, k + 32, lane);
#pragma unroll
        for (int t = 0; t < 4; ++t) bA[t] = load_b32x16(wr, Dd, t * 16, k + 32, lane);
#pragma unroll
        for (int t = 0; t < 4; ++t) acc[t] = wmma_bf16(aB, bB[t], acc[t]);
    }
    {   // tail: steps k (into B) and the one already in A
        v16bf aB = load_a16x32(xbf, Dd, row0, k, lane);
        v16bf bB[4];
#pragma unroll
        for (int t = 0; t < 4; ++t) bB[t] = load_b32x16(wr, Dd, t * 16, k, lane);
#pragma unroll
        for (int t = 0; t < 4; ++t) acc[t] = wmma_bf16(aA, bA[t], acc[t]);
#pragma unroll
        for (int t = 0; t < 4; ++t) acc[t] = wmma_bf16(aB, bB[t], acc[t]);
    }

    int half = lane >> 4, r = lane & 15;
#pragma unroll
    for (int v = 0; v < 8; ++v) {  // softmax over 64 g (4 tiles x 16 lanes/half)
        float a0 = acc[0][v] * 0.125f, a1 = acc[1][v] * 0.125f;
        float a2 = acc[2][v] * 0.125f, a3 = acc[3][v] * 0.125f;
        float m = fmaxf(fmaxf(a0, a1), fmaxf(a2, a3));
#pragma unroll
        for (int ms = 1; ms < 16; ms <<= 1) m = fmaxf(m, __shfl_xor(m, ms, 32));
        float e0 = __expf(a0 - m), e1 = __expf(a1 - m), e2 = __expf(a2 - m), e3 = __expf(a3 - m);
        float s = e0 + e1 + e2 + e3;
#pragma unroll
        for (int ms = 1; ms < 16; ms <<= 1) s += __shfl_xor(s, ms, 32);
        float inv = 1.0f / s;
        acc[0][v] = e0 * inv; acc[1][v] = e1 * inv; acc[2][v] = e2 * inv; acc[3][v] = e3 * inv;
    }
#pragma unroll
    for (int t = 0; t < 4; ++t) {
        int g = t * 16 + r;
        Pack8 pk;
#pragma unroll
        for (int v = 0; v < 8; ++v) pk.s[v] = f2bf(acc[t][v]);
        *(uint4*)(Rt + (size_t)(b * Gg + g) * Nn + n0 + 8 * half) = pk.q;
    }
}

// ------------------------------------------------- centroid aggregation ----
// C(b,g,d) = sum_n Rt(b,g,n) * xT(b,d,n)  (K=1600, 50 steps), ping-pong
__global__ __launch_bounds__(128) void centroid_kernel(const u16* __restrict__ Rt,
                                                       const u16* __restrict__ xT,
                                                       float* __restrict__ Cpre) {
    int lane = threadIdx.x & 31, wave = threadIdx.x >> 5;
    int b = blockIdx.x / 12, dt = blockIdx.x % 12;
    const u16* A = Rt + (size_t)b * Gg * Nn;
    const u16* Bm = xT + (size_t)b * Dd * Nn;
    int g0 = wave * 16, d0 = dt * 64;
    const v8f vz = {0.f,0.f,0.f,0.f,0.f,0.f,0.f,0.f};
    v8f acc[4] = {vz, vz, vz, vz};

    v16bf aA = load_a16x32(A, Nn, g0, 0, lane);
    v16bf bA[4];
#pragma unroll
    for (int t = 0; t < 4; ++t) bA[t] = load_b32x16(Bm, Nn, d0 + t * 16, 0, lane);
    int k = 32;
    for (; k < Nn - 32; k += 64) {
        v16bf aB = load_a16x32(A, Nn, g0, k, lane);
        v16bf bB[4];
#pragma unroll
        for (int t = 0; t < 4; ++t) bB[t] = load_b32x16(Bm, Nn, d0 + t * 16, k, lane);
#pragma unroll
        for (int t = 0; t < 4; ++t) acc[t] = wmma_bf16(aA, bA[t], acc[t]);
        aA = load_a16x32(A, Nn, g0, k + 32, lane);
#pragma unroll
        for (int t = 0; t < 4; ++t) bA[t] = load_b32x16(Bm, Nn, d0 + t * 16, k + 32, lane);
#pragma unroll
        for (int t = 0; t < 4; ++t) acc[t] = wmma_bf16(aB, bB[t], acc[t]);
    }
    {
        v16bf aB = load_a16x32(A, Nn, g0, k, lane);
        v16bf bB[4];
#pragma unroll
        for (int t = 0; t < 4; ++t) bB[t] = load_b32x16(Bm, Nn, d0 + t * 16, k, lane);
#pragma unroll
        for (int t = 0; t < 4; ++t) acc[t] = wmma_bf16(aA, bA[t], acc[t]);
#pragma unroll
        for (int t = 0; t < 4; ++t) acc[t] = wmma_bf16(aB, bB[t], acc[t]);
    }

    int half = lane >> 4, r = lane & 15;
#pragma unroll
    for (int t = 0; t < 4; ++t)
#pragma unroll
        for (int v = 0; v < 8; ++v)
            Cpre[((size_t)b * Gg + g0 + v + 8 * half) * Dd + d0 + t * 16 + r] = acc[t][v];
}

// ------------------------------------------------------------ GEMM ---------
// Y = X @ W^T (+bias). X:(M,K) bf16 rowmajor, W:(Nc,K) bf16 rowmajor.
// 32x64 tile per wave, ping-pong double-buffered (K/32 even).
// MODE 0: f32 rowmajor out. MODE 1: bf16 rowmajor out.
// MODE 2: bf16 transposed per 64-row batch: out[b][d][g], b=m/64, g=m%64.
template <int MODE>
__global__ __launch_bounds__(128) void gemm_xwT(const u16* __restrict__ X,
                                                const u16* __restrict__ W,
                                                const float* __restrict__ bias,
                                                void* __restrict__ Y,
                                                int M, int K, int Nc) {
    int lane = threadIdx.x & 31, wave = threadIdx.x >> 5;
    int nt64 = Nc >> 6;
    int mt = blockIdx.x / nt64, nt = blockIdx.x % nt64;
    int m0 = mt * 128 + wave * 32, n0 = nt * 64;
    const v8f vz = {0.f,0.f,0.f,0.f,0.f,0.f,0.f,0.f};
    v8f acc[2][4] = {{vz, vz, vz, vz}, {vz, vz, vz, vz}};

    v16bf a0A = load_a16x32(X, K, m0, 0, lane);
    v16bf a1A = load_a16x32(X, K, m0 + 16, 0, lane);
    v16bf bA[4];
#pragma unroll
    for (int t = 0; t < 4; ++t) bA[t] = load_b32x16(W, K, n0 + t * 16, 0, lane);

    int k = 32;
    for (; k < K - 32; k += 64) {
        v16bf a0B = load_a16x32(X, K, m0, k, lane);
        v16bf a1B = load_a16x32(X, K, m0 + 16, k, lane);
        v16bf bB[4];
#pragma unroll
        for (int t = 0; t < 4; ++t) bB[t] = load_b32x16(W, K, n0 + t * 16, k, lane);
#pragma unroll
        for (int t = 0; t < 4; ++t) acc[0][t] = wmma_bf16(a0A, bA[t], acc[0][t]);
#pragma unroll
        for (int t = 0; t < 4; ++t) acc[1][t] = wmma_bf16(a1A, bA[t], acc[1][t]);
        a0A = load_a16x32(X, K, m0, k + 32, lane);
        a1A = load_a16x32(X, K, m0 + 16, k + 32, lane);
#pragma unroll
        for (int t = 0; t < 4; ++t) bA[t] = load_b32x16(W, K, n0 + t * 16, k + 32, lane);
#pragma unroll
        for (int t = 0; t < 4; ++t) acc[0][t] = wmma_bf16(a0B, bB[t], acc[0][t]);
#pragma unroll
        for (int t = 0; t < 4; ++t) acc[1][t] = wmma_bf16(a1B, bB[t], acc[1][t]);
    }
    {
        v16bf a0B = load_a16x32(X, K, m0, k, lane);
        v16bf a1B = load_a16x32(X, K, m0 + 16, k, lane);
        v16bf bB[4];
#pragma unroll
        for (int t = 0; t < 4; ++t) bB[t] = load_b32x16(W, K, n0 + t * 16, k, lane);
#pragma unroll
        for (int t = 0; t < 4; ++t) acc[0][t] = wmma_bf16(a0A, bA[t], acc[0][t]);
#pragma unroll
        for (int t = 0; t < 4; ++t) acc[1][t] = wmma_bf16(a1A, bA[t], acc[1][t]);
#pragma unroll
        for (int t = 0; t < 4; ++t) acc[0][t] = wmma_bf16(a0B, bB[t], acc[0][t]);
#pragma unroll
        for (int t = 0; t < 4; ++t) acc[1][t] = wmma_bf16(a1B, bB[t], acc[1][t]);
    }

    int half = lane >> 4, r = lane & 15;
    if (MODE == 0) {
        float* out = (float*)Y;
#pragma unroll
        for (int ms = 0; ms < 2; ++ms)
#pragma unroll
            for (int t = 0; t < 4; ++t) {
                int col = n0 + t * 16 + r;
                float bv = bias ? bias[col] : 0.0f;
#pragma unroll
                for (int v = 0; v < 8; ++v)
                    out[(size_t)(m0 + 16 * ms + v + 8 * half) * Nc + col] = acc[ms][t][v] + bv;
            }
    } else if (MODE == 1) {
        u16* out = (u16*)Y;
#pragma unroll
        for (int ms = 0; ms < 2; ++ms)
#pragma unroll
            for (int t = 0; t < 4; ++t) {
                int col = n0 + t * 16 + r;
                float bv = bias ? bias[col] : 0.0f;
#pragma unroll
                for (int v = 0; v < 8; ++v)
                    out[(size_t)(m0 + 16 * ms + v + 8 * half) * Nc + col] = f2bf(acc[ms][t][v] + bv);
            }
    } else {
        u16* out = (u16*)Y;
        int bb2 = m0 >> 6;           // both subtiles stay in the same 64-row batch
#pragma unroll
        for (int ms = 0; ms < 2; ++ms) {
            int gbase = ((m0 + 16 * ms) & 63) + 8 * half;
#pragma unroll
            for (int t = 0; t < 4; ++t) {
                int d = n0 + t * 16 + r;
                Pack8 pk;
#pragma unroll
                for (int v = 0; v < 8; ++v) pk.s[v] = f2bf(acc[ms][t][v]);
                *(uint4*)(out + (size_t)bb2 * Nc * 64 + (size_t)d * 64 + gbase) = pk.q;
            }
        }
    }
}

// ------------------------------------------------------- LayerNorm ---------
__global__ void ln_c_kernel(const float* __restrict__ in, const float* __restrict__ gamma,
                            const float* __restrict__ beta, u16* __restrict__ outb) {
    __shared__ float red[256];
    int row = blockIdx.x, tid = threadIdx.x;
    const float* p = in + (size_t)row * Dd;
    float v0 = p[tid], v1 = p[tid + 256], v2 = p[tid + 512];
    red[tid] = v0 + v1 + v2; __syncthreads();
    for (int st = 128; st > 0; st >>= 1) { if (tid < st) red[tid] += red[tid + st]; __syncthreads(); }
    float mean = red[0] * (1.0f / 768.0f);
    __syncthreads();
    float d0 = v0 - mean, d1 = v1 - mean, d2 = v2 - mean;
    red[tid] = d0 * d0 + d1 * d1 + d2 * d2; __syncthreads();
    for (int st = 128; st > 0; st >>= 1) { if (tid < st) red[tid] += red[tid + st]; __syncthreads(); }
    float inv = rsqrtf(red[0] * (1.0f / 768.0f) + 1e-5f);
    u16* o = outb + (size_t)row * Dd;
    o[tid]       = f2bf(d0 * inv * gamma[tid]       + beta[tid]);
    o[tid + 256] = f2bf(d1 * inv * gamma[tid + 256] + beta[tid + 256]);
    o[tid + 512] = f2bf(d2 * inv * gamma[tid + 512] + beta[tid + 512]);
}

__global__ void final_ln_kernel(const float* __restrict__ x, const float* __restrict__ O,
                                const float* __restrict__ gamma, const float* __restrict__ beta,
                                float* __restrict__ out) {
    __shared__ float red[256];
    int row = blockIdx.x, tid = threadIdx.x;
    size_t base = (size_t)row * Dd;
    float v0 = x[base + tid]       + O[base + tid];
    float v1 = x[base + tid + 256] + O[base + tid + 256];
    float v2 = x[base + tid + 512] + O[base + tid + 512];
    red[tid] = v0 + v1 + v2; __syncthreads();
    for (int st = 128; st > 0; st >>= 1) { if (tid < st) red[tid] += red[tid + st]; __syncthreads(); }
    float mean = red[0] * (1.0f / 768.0f);
    __syncthreads();
    float d0 = v0 - mean, d1 = v1 - mean, d2 = v2 - mean;
    red[tid] = d0 * d0 + d1 * d1 + d2 * d2; __syncthreads();
    for (int st = 128; st > 0; st >>= 1) { if (tid < st) red[tid] += red[tid + st]; __syncthreads(); }
    float inv = rsqrtf(red[0] * (1.0f / 768.0f) + 1e-5f);
    out[base + tid]       = d0 * inv * gamma[tid]       + beta[tid];
    out[base + tid + 256] = d1 * inv * gamma[tid + 256] + beta[tid + 256];
    out[base + tid + 512] = d2 * inv * gamma[tid + 512] + beta[tid + 512];
}

// ------------------------------------------ centroid self-attention --------
__global__ void centroid_attn_kernel(const u16* __restrict__ qkv, u16* __restrict__ attnC) {
    __shared__ u16 qs[64 * 96], ks_[64 * 96], vs[64 * 96];
    __shared__ float S[64 * 64];
    int tid = threadIdx.x;
    int b = blockIdx.x >> 3, h = blockIdx.x & 7;
    for (int i = tid; i < 64 * 96; i += 256) {
        int g = i / 96, e = i % 96;
        size_t base = (size_t)(b * Gg + g) * (3 * Dd) + h * HDd + e;
        qs[i] = qkv[base]; ks_[i] = qkv[base + Dd]; vs[i] = qkv[base + 2 * Dd];
    }
    __syncthreads();
    int rr = tid >> 2, quad = tid & 3;
    for (int jj = 0; jj < 16; ++jj) {
        int j = quad * 16 + jj;
        float s = 0.f;
        for (int e = 0; e < 96; ++e) s += bf2f(qs[rr * 96 + e]) * bf2f(ks_[j * 96 + e]);
        S[rr * 64 + j] = s * 0.102062072615966f;
    }
    __syncthreads();
    if (tid < 64) {
        float m = -1e30f;
        for (int j = 0; j < 64; ++j) m = fmaxf(m, S[tid * 64 + j]);
        float s = 0.f;
        for (int j = 0; j < 64; ++j) { float e = __expf(S[tid * 64 + j] - m); S[tid * 64 + j] = e; s += e; }
        float inv = 1.0f / s;
        for (int j = 0; j < 64; ++j) S[tid * 64 + j] *= inv;
    }
    __syncthreads();
    for (int i = tid; i < 64 * 96; i += 256) {
        int g = i / 96, e = i % 96;
        float s = 0.f;
        for (int j = 0; j < 64; ++j) s += S[g * 64 + j] * bf2f(vs[j * 96 + e]);
        attnC[(size_t)(b * Gg + g) * Dd + h * HDd + e] = f2bf(s);
    }
}

// ------------------------------------------------- N x G cross-attn --------
__global__ __launch_bounds__(128) void cross_attn_kernel(const u16* __restrict__ Q,
                                                         const u16* __restrict__ Kc,
                                                         const u16* __restrict__ Vt,
                                                         u16* __restrict__ ctx) {
    __shared__ __align__(16) u16 P[4][16 * 64];
    int lane = threadIdx.x & 31, wave = threadIdx.x >> 5;
    int idx = blockIdx.x * 4 + wave;
    int nt = idx % 100; int bh = idx / 100;
    int h = bh & 7, b = bh >> 3;
    int row0 = b * Nn + nt * 16;
    const u16* Kb = Kc + (size_t)b * Gg * Dd + h * HDd;
    const v8f vz = {0.f,0.f,0.f,0.f,0.f,0.f,0.f,0.f};
    v8f acc[4] = {vz, vz, vz, vz};
    for (int k0 = 0; k0 < HDd; k0 += 32) {
        v16bf a = load_a16x32(Q, Dd, row0, h * HDd + k0, lane);
#pragma unroll
        for (int t = 0; t < 4; ++t)
            acc[t] = wmma_bf16(a, load_b32x16(Kb, Dd, t * 16, k0, lane), acc[t]);
    }
    int half = lane >> 4, r = lane & 15;
    const float sc = 0.102062072615966f;  // 96^-0.5
#pragma unroll
    for (int v = 0; v < 8; ++v) {
        float a0 = acc[0][v] * sc, a1 = acc[1][v] * sc, a2 = acc[2][v] * sc, a3 = acc[3][v] * sc;
        float m = fmaxf(fmaxf(a0, a1), fmaxf(a2, a3));
#pragma unroll
        for (int ms = 1; ms < 16; ms <<= 1) m = fmaxf(m, __shfl_xor(m, ms, 32));
        float e0 = __expf(a0 - m), e1 = __expf(a1 - m), e2 = __expf(a2 - m), e3 = __expf(a3 - m);
        float s = e0 + e1 + e2 + e3;
#pragma unroll
        for (int ms = 1; ms < 16; ms <<= 1) s += __shfl_xor(s, ms, 32);
        float inv = 1.0f / s;
        acc[0][v] = e0 * inv; acc[1][v] = e1 * inv; acc[2][v] = e2 * inv; acc[3][v] = e3 * inv;
    }
    // re-layout P (C-fragment) -> A-fragment via LDS
#pragma unroll
    for (int t = 0; t < 4; ++t)
#pragma unroll
        for (int v = 0; v < 8; ++v)
            P[wave][(v + 8 * half) * 64 + t * 16 + r] = f2bf(acc[t][v]);
    __syncthreads();
    v8f o[6] = {vz, vz, vz, vz, vz, vz};
    const u16* pp = &P[wave][0] + r * 64;
    const u16* Vb = Vt + (size_t)b * Dd * Gg + (size_t)h * HDd * Gg;
    for (int g0 = 0; g0 < 64; g0 += 32) {
        BFrag fa;
        fa.u[0] = *(const uint4*)(pp + g0 + half * 8);
        fa.u[1] = *(const uint4*)(pp + g0 + 16 + half * 8);
#pragma unroll
        for (int et = 0; et < 6; ++et)
            o[et] = wmma_bf16(fa.v, load_b32x16(Vb, Gg, et * 16, g0, lane), o[et]);
    }
#pragma unroll
    for (int et = 0; et < 6; ++et)
#pragma unroll
        for (int v = 0; v < 8; ++v)
            ctx[(size_t)(row0 + v + 8 * half) * Dd + h * HDd + et * 16 + r] = f2bf(o[et][v]);
}

// ---------------------------------------------------------------------------
extern "C" void kernel_launch(void* const* d_in, const int* in_sizes, int n_in,
                              void* d_out, int out_size, void* d_ws, size_t ws_size,
                              hipStream_t stream) {
    const float* x      = (const float*)d_in[0];
    const float* W_r    = (const float*)d_in[1];
    const float* Wq     = (const float*)d_in[2];
    const float* Wk     = (const float*)d_in[3];
    const float* Wv     = (const float*)d_in[4];
    const float* Wo     = (const float*)d_in[5];
    const float* min_w  = (const float*)d_in[6];
    const float* min_b  = (const float*)d_in[7];
    const float* mout_w = (const float*)d_in[8];
    const float* mout_b = (const float*)d_in[9];
    const float* gc     = (const float*)d_in[10];
    const float* bc     = (const float*)d_in[11];
    const float* gx     = (const float*)d_in[12];
    const float* bx     = (const float*)d_in[13];
    float* out = (float*)d_out;

    char* ws = (char*)d_ws;
    size_t off = 0;
    auto take = [&](size_t bytes) -> char* {
        char* p = ws + off;
        off += (bytes + 255) & ~(size_t)255;
        return p;
    };
    u16*   xbf   = (u16*)  take((size_t)Bb * Nn * Dd * 2);   // also reused as ctx
    u16*   xT    = (u16*)  take((size_t)Bb * Dd * Nn * 2);   // also reused as Q
    u16*   wrb   = (u16*)  take((size_t)Gg * Dd * 2);
    u16*   wqb   = (u16*)  take((size_t)Dd * Dd * 2);
    u16*   wkb   = (u16*)  take((size_t)Dd * Dd * 2);
    u16*   wvb   = (u16*)  take((size_t)Dd * Dd * 2);
    u16*   wob   = (u16*)  take((size_t)Dd * Dd * 2);
    u16*   winb  = (u16*)  take((size_t)3 * Dd * Dd * 2);
    u16*   woutb = (u16*)  take((size_t)Dd * Dd * 2);
    u16*   Rt    = (u16*)  take((size_t)Bb * Gg * Nn * 2);
    float* Cpre  = (float*)take((size_t)Bb * Gg * Dd * 4);
    u16*   Cln   = (u16*)  take((size_t)Bb * Gg * Dd * 2);
    u16*   qkv   = (u16*)  take((size_t)Bb * Gg * 3 * Dd * 2);
    u16*   attnC = (u16*)  take((size_t)Bb * Gg * Dd * 2);
    u16*   C2    = (u16*)  take((size_t)Bb * Gg * Dd * 2);
    u16*   Kc    = (u16*)  take((size_t)Bb * Gg * Dd * 2);
    u16*   Vt    = (u16*)  take((size_t)Bb * Dd * Gg * 2);
    float* Obuf  = (float*)take((size_t)Bb * Nn * Dd * 4);
    u16* ctx = xbf;   // x_bf16 dead after Q projection
    u16* Qb  = xT;    // xT dead after centroid aggregation

    // 1) cast x -> bf16 row-major + transposed bf16
    prep_x_kernel<<<Bb * 50 * 24, 256, 0, stream>>>(x, xbf, xT);
    // 2) cast weights to bf16
    cast_f32_bf16<<<(Gg * Dd + 255) / 256, 256, 0, stream>>>(W_r, wrb, Gg * Dd);
    cast_f32_bf16<<<(Dd * Dd + 255) / 256, 256, 0, stream>>>(Wq, wqb, Dd * Dd);
    cast_f32_bf16<<<(Dd * Dd + 255) / 256, 256, 0, stream>>>(Wk, wkb, Dd * Dd);
    cast_f32_bf16<<<(Dd * Dd + 255) / 256, 256, 0, stream>>>(Wv, wvb, Dd * Dd);
    cast_f32_bf16<<<(Dd * Dd + 255) / 256, 256, 0, stream>>>(Wo, wob, Dd * Dd);
    cast_f32_bf16<<<(3 * Dd * Dd + 255) / 256, 256, 0, stream>>>(min_w, winb, 3 * Dd * Dd);
    cast_f32_bf16<<<(Dd * Dd + 255) / 256, 256, 0, stream>>>(mout_w, woutb, Dd * Dd);
    // 3) routing softmax  (WMMA + fused softmax), writes R^T bf16
    routing_kernel<<<Bb * Nn / 64, 128, 0, stream>>>(xbf, wrb, Rt);
    // 4) centroid aggregation C = R^T x  (WMMA, K=1600)
    centroid_kernel<<<Bb * 12, 128, 0, stream>>>(Rt, xT, Cpre);
    // 5) LayerNorm(C) -> bf16
    ln_c_kernel<<<Bb * Gg, 256, 0, stream>>>(Cpre, gc, bc, Cln);
    // 6) qkv = C_ln @ mha_in_w^T + b   (WMMA)
    gemm_xwT<1><<<(Bb * Gg / 128) * (3 * Dd / 64), 128, 0, stream>>>(Cln, winb, min_b, qkv,
                                                                     Bb * Gg, Dd, 3 * Dd);
    // 7) centroid self-attention (64x64 per b,h)
    centroid_attn_kernel<<<Bb * Hh, 256, 0, stream>>>(qkv, attnC);
    // 8) C2 = attnC @ mha_out_w^T + b  (WMMA)
    gemm_xwT<1><<<(Bb * Gg / 128) * (Dd / 64), 128, 0, stream>>>(attnC, woutb, mout_b, C2,
                                                                 Bb * Gg, Dd, Dd);
    // 9) Kc = C2 @ Wk^T ; Vt = transpose_per_batch(C2 @ Wv^T)
    gemm_xwT<1><<<(Bb * Gg / 128) * (Dd / 64), 128, 0, stream>>>(C2, wkb, nullptr, Kc,
                                                                 Bb * Gg, Dd, Dd);
    gemm_xwT<2><<<(Bb * Gg / 128) * (Dd / 64), 128, 0, stream>>>(C2, wvb, nullptr, Vt,
                                                                 Bb * Gg, Dd, Dd);
    // 10) Q = x @ Wq^T  (big WMMA GEMM, 60 GFLOP)
    gemm_xwT<1><<<(Bb * Nn / 128) * (Dd / 64), 128, 0, stream>>>(xbf, wqb, nullptr, Qb,
                                                                 Bb * Nn, Dd, Dd);
    // 11) cross attention: S=QK^T -> softmax -> P@V (both WMMA, fused)
    cross_attn_kernel<<<Bb * Hh * 100 / 4, 128, 0, stream>>>(Qb, Kc, Vt, ctx);
    // 12) O = ctx @ Wo^T  (big WMMA GEMM, f32 out)
    gemm_xwT<0><<<(Bb * Nn / 128) * (Dd / 64), 128, 0, stream>>>(ctx, wob, nullptr, Obuf,
                                                                 Bb * Nn, Dd, Dd);
    // 13) out = LN(x + O)
    final_ln_kernel<<<Bb * Nn, 256, 0, stream>>>(x, Obuf, gx, bx, out);
}
